// GatedGraphNN_2602750181410
// MI455X (gfx1250) — matrix operand order repaired and checked by reading, not compiled, not run on hardware
//
#include <hip/hip_runtime.h>

// GGNN: B=4, N=4096, D=64, L=3.
// Heavy op: per layer two dense [N,N]x[N,64] matmuls (A@x, A^T@x) -> WMMA bf16.
// A is binary -> exact in bf16; accumulate in f32 (v_wmma_f32_16x16x32_bf16).

#define GB 4
#define GN 4096
#define GD 64
#define GL 3

typedef __bf16 bf16;
typedef __attribute__((ext_vector_type(16))) bf16  v16bf;
typedef __attribute__((ext_vector_type(8)))  bf16  v8bf;
typedef __attribute__((ext_vector_type(8)))  float v8f;

union V16U { v16bf v; v8bf h[2]; };

static __device__ __forceinline__ v8f wmma_bf16f32(v16bf a, v16bf b, v8f c) {
  // D = A(16x32 bf16) * B(32x16 bf16) + C(16x16 f32)
  return __builtin_amdgcn_wmma_f32_16x16x32_bf16(false, a, false, b, (short)0, c, false, false);
}
static __device__ __forceinline__ float sigmoidf_(float x) { return 1.0f / (1.0f + __expf(-x)); }

// ---------------------------------------------------------------------------
// One-time preprocessing
// ---------------------------------------------------------------------------

// norms start at 1.0 (the "+1" of degree+1); prep_adj atomically adds degrees.
__global__ __launch_bounds__(256) void norm_init_kernel(float* __restrict__ a,
                                                        float* __restrict__ b) {
  int i = blockIdx.x * 256 + threadIdx.x;
  a[i] = 1.0f;
  b[i] = 1.0f;
}

// Single pass over adj (fp32, read once per element):
//   - Abf  = bf16(adj)            (row-major, coalesced)
//   - ATbf = bf16(adj^T)          (via LDS tile, coalesced)
//   - normA  += row sums, normAT += col sums (exact: 0/1 integers -> fp32
//     atomic adds are order-independent and bitwise deterministic)
__global__ __launch_bounds__(256) void prep_adj_kernel(const float* __restrict__ A,
                                                       bf16* __restrict__ Ab,
                                                       bf16* __restrict__ ATb,
                                                       float* __restrict__ normA,
                                                       float* __restrict__ normAT) {
  __shared__ float t[64][65];
  const int b = blockIdx.z, ti = blockIdx.y, tj = blockIdx.x;
  #pragma unroll
  for (int p = 0; p < 16; ++p) {
    int idx = threadIdx.x + p * 256;
    int r = idx >> 6, c = idx & 63;
    float v = A[((size_t)b * GN + ti * 64 + r) * GN + tj * 64 + c];
    t[r][c] = v;
    Ab[((size_t)b * GN + ti * 64 + r) * GN + tj * 64 + c] = (bf16)v;
  }
  __syncthreads();
  #pragma unroll
  for (int p = 0; p < 16; ++p) {
    int idx = threadIdx.x + p * 256;
    int r = idx >> 6, c = idx & 63;   // output row (= col of A), output col (= row of A)
    ATb[((size_t)b * GN + tj * 64 + r) * GN + ti * 64 + c] = (bf16)t[c][r];
  }
  if (threadIdx.x < 64) {            // row sums of this tile -> normA
    int r = threadIdx.x;
    float s = 0.0f;
    #pragma unroll 8
    for (int c = 0; c < 64; ++c) s += t[r][c];
    atomicAdd(&normA[(size_t)b * GN + ti * 64 + r], s);
  } else if (threadIdx.x < 128) {    // col sums of this tile -> normAT
    int c = threadIdx.x - 64;
    float s = 0.0f;
    #pragma unroll 8
    for (int r = 0; r < 64; ++r) s += t[r][c];
    atomicAdd(&normAT[(size_t)b * GN + tj * 64 + c], s);
  }
}

// W [K,E] fp32 -> WT [E,K] bf16 (tiny matrices).
__global__ __launch_bounds__(256) void transpose_conv_kernel(const float* __restrict__ W,
                                                             bf16* __restrict__ WT,
                                                             int K, int E) {
  int i = blockIdx.x * 256 + threadIdx.x;
  if (i < K * E) {
    int k = i / E, e = i % E;
    WT[e * K + k] = (bf16)W[i];
  }
}

// h fp32 [B,N,64] -> hT bf16 [B,64,N] (64x64 LDS tiles).
__global__ __launch_bounds__(256) void xT_kernel(const float* __restrict__ h,
                                                 bf16* __restrict__ hT) {
  __shared__ float t[64][65];
  const int b = blockIdx.y, n0 = blockIdx.x * 64;
  #pragma unroll
  for (int p = 0; p < 16; ++p) {
    int idx = threadIdx.x + p * 256;
    int r = idx >> 6, c = idx & 63;            // node-local, dim
    t[r][c] = h[((size_t)b * GN + n0 + r) * GD + c];
  }
  __syncthreads();
  #pragma unroll
  for (int p = 0; p < 16; ++p) {
    int idx = threadIdx.x + p * 256;
    int d = idx >> 6, nl = idx & 63;
    hT[((size_t)b * GD + d) * GN + n0 + nl] = (bf16)t[nl][d];
  }
}

// ---------------------------------------------------------------------------
// Heavy kernel: out = relu( ((A@x + x)/norm) @ Wm + bm )
// One wave per 16-row strip; 4 f32 accumulators cover D=64; K loop over 4096.
// Second GEMM (x Wm, K=64) done via LDS bf16 staging -> 8 more WMMAs.
// ---------------------------------------------------------------------------
__global__ __launch_bounds__(128) void msg_mm_kernel(
    const bf16*  __restrict__ Abf,   // [B][N][N] bf16 (already direction-correct)
    const bf16*  __restrict__ hT,    // [B][64][N] bf16
    const float* __restrict__ h,     // [B][N][64] fp32 (residual +x)
    const float* __restrict__ norm,  // [B][N]
    const bf16*  __restrict__ WmT,   // [64][64] bf16, [e][k]
    const float* __restrict__ bm,    // [64]
    float*       __restrict__ out)   // [B][N][64]
{
  __shared__ bf16 stage[4][16 * 64];
  const int w     = threadIdx.x >> 5;
  const int lane  = threadIdx.x & 31;
  const int strip = blockIdx.x * 4 + w;          // 0 .. B*N/16-1
  const int b     = strip / (GN / 16);
  const int row0  = (strip % (GN / 16)) * 16;
  const int m  = lane & 15;                      // A: row-in-tile / B,C: col-in-tile
  const int kh = lane >> 4;

  const bf16* arow  = Abf + ((size_t)b * GN + row0 + m) * GN + kh * 16;
  const bf16* bbase = hT + ((size_t)b * GD + m) * GN + kh * 16;

  v8f acc[4] = {};
  for (int kb = 0; kb < GN; kb += 32) {
    // stream-ahead hint on the HBM-bound A stream: one new 128B line per row
    // every 64 elements -> global_prefetch_b8, no LOADcnt cost
    if ((kb & 63) == 0) __builtin_prefetch(arow + kb + 2048, 0, 0);
    V16U a;
    a.h[0] = *(const v8bf*)(arow + kb);
    a.h[1] = *(const v8bf*)(arow + kb + 8);
    #pragma unroll
    for (int c = 0; c < 4; ++c) {
      v16bf bv = *(const v16bf*)(bbase + (size_t)c * 16 * GN + kb);
      acc[c] = wmma_bf16f32(a.v, bv, acc[c]);
    }
  }

  // agg = (T + x)/norm at C-layout positions; stash bf16 row-major in LDS.
  bf16* st = stage[w];
  float nrm[8];
  #pragma unroll
  for (int r = 0; r < 8; ++r) nrm[r] = norm[(size_t)b * GN + row0 + r + 8 * kh];
  #pragma unroll
  for (int c = 0; c < 4; ++c) {
    #pragma unroll
    for (int r = 0; r < 8; ++r) {
      int lm = r + 8 * kh, col = c * 16 + m;
      float xv = h[((size_t)b * GN + row0 + lm) * GD + col];
      st[lm * 64 + col] = (bf16)((acc[c][r] + xv) / nrm[r]);
    }
  }
  __syncthreads();

  // out = relu(agg @ Wm + bm): A tiles from LDS, B tiles from WmT.
  V16U a0, a1;
  a0.h[0] = *(const v8bf*)(st + m * 64 + kh * 16);
  a0.h[1] = *(const v8bf*)(st + m * 64 + kh * 16 + 8);
  a1.h[0] = *(const v8bf*)(st + m * 64 + 32 + kh * 16);
  a1.h[1] = *(const v8bf*)(st + m * 64 + 32 + kh * 16 + 8);
  #pragma unroll
  for (int c = 0; c < 4; ++c) {
    v16bf b0 = *(const v16bf*)(WmT + (c * 16 + m) * 64 + kh * 16);
    v16bf b1 = *(const v16bf*)(WmT + (c * 16 + m) * 64 + 32 + kh * 16);
    v8f o = {};
    o = wmma_bf16f32(a0.v, b0, o);
    o = wmma_bf16f32(a1.v, b1, o);
    float bias = bm[c * 16 + m];
    #pragma unroll
    for (int r = 0; r < 8; ++r) {
      float v = o[r] + bias;
      out[((size_t)b * GN + row0 + r + 8 * kh) * GD + c * 16 + m] = v > 0.0f ? v : 0.0f;
    }
  }
}

// ---------------------------------------------------------------------------
// Gated fusion + GRU step, all four small GEMMs as WMMA via LDS bf16 staging.
// One wave per 16 nodes; 2 waves per block.
// ---------------------------------------------------------------------------
__global__ __launch_bounds__(64) void fuse_gru_kernel(
    const float* __restrict__ h,   const float* __restrict__ fw, const float* __restrict__ bw,
    const bf16*  __restrict__ WzfT, const float* __restrict__ bzf,
    const bf16*  __restrict__ WzT, const bf16* __restrict__ WrT, const bf16* __restrict__ WtT,
    float*       __restrict__ out)
{
  __shared__ float lds_fw[2][16 * 64];
  __shared__ float lds_bw[2][16 * 64];
  __shared__ bf16  lds_st[2][16 * 256];
  const int w    = threadIdx.x >> 5;
  const int lane = threadIdx.x & 31;
  const size_t rowbase = ((size_t)blockIdx.x * 2 + w) * 16;  // flat node index over B*N
  const int m = lane & 15, kh = lane >> 4;
  float* Lf = lds_fw[w];
  float* Lb = lds_bw[w];
  bf16*  St = lds_st[w];

  // phase 1: fw/bw rows -> LDS fp32
  #pragma unroll 4
  for (int j = 0; j < 32; ++j) {
    int idx = lane + j * 32;
    Lf[idx] = fw[rowbase * GD + idx];
    Lb[idx] = bw[rowbase * GD + idx];
  }
  __syncthreads();

  // phase 2: fus_in = [fw, bw, fw*bw, fw-bw] -> LDS bf16 [16][256]
  for (int j = 0; j < 128; ++j) {
    int idx = lane + j * 32;
    int rr = idx >> 8, kk = idx & 255;
    float f = Lf[rr * 64 + (kk & 63)];
    float g = Lb[rr * 64 + (kk & 63)];
    float v = (kk < 64) ? f : (kk < 128) ? g : (kk < 192) ? f * g : f - g;
    St[idx] = (bf16)v;
  }
  __syncthreads();

  // phase 3: z = sigmoid(fus_in @ Wzf + bzf)   (K=256 -> 8 ksteps x 4 tiles)
  v8f zacc[4] = {};
  for (int kb = 0; kb < 256; kb += 32) {
    V16U a;
    a.h[0] = *(const v8bf*)(St + m * 256 + kb + kh * 16);
    a.h[1] = *(const v8bf*)(St + m * 256 + kb + 8 + kh * 16);
    #pragma unroll
    for (int c = 0; c < 4; ++c) {
      v16bf bv = *(const v16bf*)(WzfT + (c * 16 + m) * 256 + kb + kh * 16);
      zacc[c] = wmma_bf16f32(a.v, bv, zacc[c]);
    }
  }
  __syncthreads();

  // phase 4: agg = (1-z)*fw + z*bw; stage hx = [agg, x] bf16 [16][128]
  float aggr[4][8];
  #pragma unroll
  for (int c = 0; c < 4; ++c) {
    float bias = bzf[c * 16 + m];
    #pragma unroll
    for (int r = 0; r < 8; ++r) {
      int lm = r + 8 * kh, col = c * 16 + m;
      float z  = sigmoidf_(zacc[c][r] + bias);
      float av = (1.0f - z) * Lf[lm * 64 + col] + z * Lb[lm * 64 + col];
      aggr[c][r] = av;
      St[lm * 128 + col] = (bf16)av;
    }
  }
  #pragma unroll 4
  for (int j = 0; j < 32; ++j) {
    int idx = lane + j * 32;
    int rr = idx >> 6, dd = idx & 63;
    St[rr * 128 + 64 + dd] = (bf16)h[(rowbase + rr) * GD + dd];
  }
  __syncthreads();

  // phase 5: zg = sigmoid(hx @ Wz), r = sigmoid(hx @ Wr)   (K=128)
  v8f ga[4] = {}, ra[4] = {};
  for (int kb = 0; kb < 128; kb += 32) {
    V16U a;
    a.h[0] = *(const v8bf*)(St + m * 128 + kb + kh * 16);
    a.h[1] = *(const v8bf*)(St + m * 128 + kb + 8 + kh * 16);
    #pragma unroll
    for (int c = 0; c < 4; ++c) {
      v16bf bz = *(const v16bf*)(WzT + (c * 16 + m) * 128 + kb + kh * 16);
      v16bf br = *(const v16bf*)(WrT + (c * 16 + m) * 128 + kb + kh * 16);
      ga[c] = wmma_bf16f32(a.v, bz, ga[c]);
      ra[c] = wmma_bf16f32(a.v, br, ra[c]);
    }
  }
  __syncthreads();

  // phase 6: rhx = [r*agg, x]: overwrite cols 0..63 (x half unchanged)
  #pragma unroll
  for (int c = 0; c < 4; ++c) {
    #pragma unroll
    for (int r = 0; r < 8; ++r) {
      int lm = r + 8 * kh, col = c * 16 + m;
      float rv = sigmoidf_(ra[c][r]);
      St[lm * 128 + col] = (bf16)(rv * aggr[c][r]);
    }
  }
  __syncthreads();

  // phase 7: t = tanh(rhx @ Wt)
  v8f ta[4] = {};
  for (int kb = 0; kb < 128; kb += 32) {
    V16U a;
    a.h[0] = *(const v8bf*)(St + m * 128 + kb + kh * 16);
    a.h[1] = *(const v8bf*)(St + m * 128 + kb + 8 + kh * 16);
    #pragma unroll
    for (int c = 0; c < 4; ++c) {
      v16bf bt = *(const v16bf*)(WtT + (c * 16 + m) * 128 + kb + kh * 16);
      ta[c] = wmma_bf16f32(a.v, bt, ta[c]);
    }
  }

  // phase 8: h_new = (1-zg)*agg + zg*t
  #pragma unroll
  for (int c = 0; c < 4; ++c) {
    #pragma unroll
    for (int r = 0; r < 8; ++r) {
      int lm = r + 8 * kh, col = c * 16 + m;
      float zg = sigmoidf_(ga[c][r]);
      float tt = tanhf(ta[c][r]);
      out[(rowbase + lm) * GD + col] = (1.0f - zg) * aggr[c][r] + zg * tt;
    }
  }
}

// ---------------------------------------------------------------------------
extern "C" void kernel_launch(void* const* d_in, const int* in_sizes, int n_in,
                              void* d_out, int out_size, void* d_ws, size_t ws_size,
                              hipStream_t stream) {
  (void)in_sizes; (void)n_in; (void)out_size; (void)ws_size;
  const float* x   = (const float*)d_in[0];
  const float* adj = (const float*)d_in[1];
  const float* Wm  = (const float*)d_in[2];
  const float* bm  = (const float*)d_in[3];
  const float* Wzf = (const float*)d_in[4];
  const float* bzf = (const float*)d_in[5];
  const float* Wz  = (const float*)d_in[6];
  const float* Wr  = (const float*)d_in[7];
  const float* Wt  = (const float*)d_in[8];
  float* out = (float*)d_out;

  // deterministic workspace carve-out (256B aligned)
  char* base = (char*)d_ws;
  size_t off = 0;
  auto take = [&](size_t bytes) -> char* {
    char* r = base + off;
    off = (off + bytes + 255) & ~(size_t)255;
    return r;
  };
  const size_t BNN = (size_t)GB * GN * GN;
  const size_t BND = (size_t)GB * GN * GD;
  bf16*  Abf    = (bf16*) take(BNN * 2);
  bf16*  ATbf   = (bf16*) take(BNN * 2);
  float* normA  = (float*)take((size_t)GB * GN * 4);
  float* normAT = (float*)take((size_t)GB * GN * 4);
  bf16*  hT     = (bf16*) take(BND * 2);
  float* bwb    = (float*)take(BND * 4);
  float* fwb    = (float*)take(BND * 4);
  float* hb0    = (float*)take(BND * 4);
  float* hb1    = (float*)take(BND * 4);
  bf16*  WmT    = (bf16*) take((size_t)GL * GD * GD * 2);
  bf16*  WzfT   = (bf16*) take((size_t)GL * GD * 4 * GD * 2);
  bf16*  WzT    = (bf16*) take((size_t)GL * GD * 2 * GD * 2);
  bf16*  WrT    = (bf16*) take((size_t)GL * GD * 2 * GD * 2);
  bf16*  WtT    = (bf16*) take((size_t)GL * GD * 2 * GD * 2);

  // one-time preprocessing: single pass over adj
  norm_init_kernel<<<GB * GN / 256, 256, 0, stream>>>(normA, normAT);
  prep_adj_kernel<<<dim3(GN / 64, GN / 64, GB), 256, 0, stream>>>(adj, Abf, ATbf, normA, normAT);
  for (int l = 0; l < GL; ++l) {
    transpose_conv_kernel<<<(64 * 64 + 255) / 256, 256, 0, stream>>>(Wm + l * 64 * 64, WmT + l * 64 * 64, 64, 64);
    transpose_conv_kernel<<<(256 * 64 + 255) / 256, 256, 0, stream>>>(Wzf + l * 256 * 64, WzfT + l * 64 * 256, 256, 64);
    transpose_conv_kernel<<<(128 * 64 + 255) / 256, 256, 0, stream>>>(Wz + l * 128 * 64, WzT + l * 64 * 128, 128, 64);
    transpose_conv_kernel<<<(128 * 64 + 255) / 256, 256, 0, stream>>>(Wr + l * 128 * 64, WrT + l * 64 * 128, 128, 64);
    transpose_conv_kernel<<<(128 * 64 + 255) / 256, 256, 0, stream>>>(Wt + l * 128 * 64, WtT + l * 64 * 128, 128, 64);
  }

  // layers
  const float* hin = x;
  float* houts[GL] = { hb0, hb1, out };
  for (int l = 0; l < GL; ++l) {
    xT_kernel<<<dim3(GN / 64, GB), 256, 0, stream>>>(hin, hT);
    msg_mm_kernel<<<GB * GN / 64, 128, 0, stream>>>(Abf, hT, hin, normA,
                                                    WmT + l * 64 * 64, bm + l * 64, bwb);
    msg_mm_kernel<<<GB * GN / 64, 128, 0, stream>>>(ATbf, hT, hin, normAT,
                                                    WmT + l * 64 * 64, bm + l * 64, fwb);
    fuse_gru_kernel<<<GB * GN / 32, 64, 0, stream>>>(hin, fwb, bwb,
                                                     WzfT + l * 64 * 256, bzf + l * 64,
                                                     WzT + l * 64 * 128, WrT + l * 64 * 128,
                                                     WtT + l * 64 * 128, houts[l]);
    hin = houts[l];
  }
}
